// DTAttention_78365973282909
// MI455X (gfx1250) — matrix operand order repaired and checked by reading, not compiled, not run on hardware
//
#include <hip/hip_runtime.h>
#include <hip/hip_bf16.h>

typedef __attribute__((ext_vector_type(16))) _Float16 v16h;
typedef __attribute__((ext_vector_type(8)))  _Float16 v8h;
typedef __attribute__((ext_vector_type(8)))  float    v8f;

#define WMMA_F16(a, b, c) \
    __builtin_amdgcn_wmma_f32_16x16x32_f16(false, (a), false, (b), (short)0, (c), false, false)

static __device__ __forceinline__ v16h cat8(v8h lo, v8h hi) {
    return __builtin_shufflevector(lo, hi, 0,1,2,3,4,5,6,7,8,9,10,11,12,13,14,15);
}
static __device__ __forceinline__ v8h ld8(const _Float16* p) {
    return *(const v8h*)p;   // 16B global/LDS load
}

static constexpr int BB   = 2;
static constexpr int LL   = 2048;
static constexpr int HID  = 2048;
static constexpr int NH   = 16;
static constexpr int NKV  = 4;
static constexpr int DD   = 128;
static constexpr float LAMBDA_INIT = 0.2f;

// ---------------------------------------------------------------------------
// WMMA GEMM, all-f16 operands, K contiguous for both:
//   C[M][N] f32 = A[M][K] f16 (row-major) * Bt[N][K] f16 (B transposed)
// 256 threads = 8 waves tiled 2(M) x 4(N); wave tile 64x32; grid (N/128, M/128)
// ---------------------------------------------------------------------------
__global__ __launch_bounds__(256) void gemm_f16t_kernel(
    const _Float16* __restrict__ A, const _Float16* __restrict__ Bt,
    float* __restrict__ C, int M, int N, int K)
{
    const int lane = threadIdx.x & 31;
    const int wave = threadIdx.x >> 5;
    const int bm   = blockIdx.y * 128 + (wave >> 2) * 64;
    const int bn   = blockIdx.x * 128 + (wave & 3) * 32;
    const int l15  = lane & 15;
    const int lhi  = lane >> 4;

    // per-lane streaming pointers (advance by 32 elements per k-step)
    const _Float16* ap0 = A  + (size_t)(bm + l15) * K + (lhi << 4);
    const _Float16* bp0 = Bt + (size_t)(bn + l15) * K + (lhi << 4);
    const size_t arow = (size_t)16 * K;   // fragment-to-fragment stride (elems)

    v8f acc[4][2] = {};

    #pragma unroll 2
    for (int k0 = 0; k0 < K; k0 += 32) {
        v16h afr[4], bfr[2];
        #pragma unroll
        for (int i = 0; i < 4; ++i) {    // A 16x32: 16 contiguous K per lane
            afr[i] = cat8(ld8(ap0 + i * arow), ld8(ap0 + i * arow + 8));
            __builtin_prefetch(ap0 + i * arow + 64, 0, 1);   // next-next k-tile
        }
        #pragma unroll
        for (int j = 0; j < 2; ++j) {    // B 32x16: 16 contiguous K per lane
            bfr[j] = cat8(ld8(bp0 + j * arow), ld8(bp0 + j * arow + 8));
            __builtin_prefetch(bp0 + j * arow + 64, 0, 1);
        }
        #pragma unroll
        for (int i = 0; i < 4; ++i)
            #pragma unroll
            for (int j = 0; j < 2; ++j)
                acc[i][j] = WMMA_F16(afr[i], bfr[j], acc[i][j]);
        ap0 += 32; bp0 += 32;
    }

    #pragma unroll
    for (int i = 0; i < 4; ++i)
        #pragma unroll
        for (int j = 0; j < 2; ++j) {
            float* cr = C + (size_t)(bm + i * 16 + (lhi << 3)) * N + bn + j * 16 + l15;
            #pragma unroll
            for (int r = 0; r < 8; ++r) cr[(size_t)r * N] = acc[i][j][r];
        }
}

// ---------------------------------------------------------------------------
// one-shot converters
// ---------------------------------------------------------------------------
// f32 -> f16 straight copy (hidden_states)
__global__ void f32_to_f16_kernel(const float* __restrict__ X, _Float16* __restrict__ Y)
{
    size_t i = (size_t)blockIdx.x * blockDim.x + threadIdx.x;
    Y[i] = (_Float16)X[i];
}

// weight transpose+convert: Wt[n][k] = (f16) W[k][n]
__global__ void wconv_t_kernel(const float* __restrict__ W, _Float16* __restrict__ Wt,
                               int K, int N)
{
    size_t idx = (size_t)blockIdx.x * blockDim.x + threadIdx.x;
    int k = (int)(idx % K);
    int n = (int)(idx / K);
    Wt[(size_t)n * K + k] = (_Float16)W[(size_t)k * N + n];
}

// RoPE + transpose: X f32 [b][l][nh][128] -> Out f16 [b][nh][l][128]
__global__ void rope_transpose_kernel(const float* __restrict__ X,
                                      const float* __restrict__ cosb,
                                      const float* __restrict__ sinb,
                                      float qscale, int nh,
                                      _Float16* __restrict__ Out)
{
    int idx = blockIdx.x * blockDim.x + threadIdx.x;
    int dd = idx & 63;
    int t  = idx >> 6;
    int hh = t % nh; t /= nh;
    int l  = t % LL;
    int b  = t / LL;

    const float* xp = X + ((size_t)(b * LL + l) * nh + hh) * DD;
    float x1 = xp[dd], x2 = xp[dd + 64];
    size_t cb = (size_t)(b * LL + l) * DD;
    float c1 = cosb[cb + dd],      s1 = sinb[cb + dd];
    float c2 = cosb[cb + dd + 64], s2 = sinb[cb + dd + 64];
    _Float16* op = Out + ((size_t)(b * nh + hh) * LL + l) * DD;
    op[dd]      = (_Float16)((x1 * c1 - x2 * s1) * qscale);
    op[dd + 64] = (_Float16)((x2 * c2 + x1 * s2) * qscale);
}

// V convert + double transpose: X f32 [b][l][NKV][128] -> Vt f16 [b][NKV][128][L]
__global__ void vconv_t_kernel(const float* __restrict__ X, _Float16* __restrict__ Vt)
{
    int idx = blockIdx.x * blockDim.x + threadIdx.x;
    int l = idx & (LL - 1);
    int t = idx >> 11;
    int d = t & (DD - 1); t >>= 7;
    int h = t & (NKV - 1);
    int b = t >> 2;
    Vt[((size_t)(b * NKV + h) * DD + d) * LL + l] =
        (_Float16)X[((size_t)(b * LL + l) * NKV + h) * DD + d];
}

// ---------------------------------------------------------------------------
// Flash differential attention. One wave = 16 query rows of one (b,h).
// Q pre-scaled by 1/sqrt(D). 32 keys per iteration. V is key-contiguous (Vt).
// ---------------------------------------------------------------------------
__global__ __launch_bounds__(256) void attn_kernel(
    const _Float16* __restrict__ Qh,   // [b][NH ][L][128]
    const _Float16* __restrict__ Kh,   // [b][NKV][L][128]
    const _Float16* __restrict__ Vt,   // [b][NKV][128][L]
    const float* __restrict__ lq1, const float* __restrict__ lk1,
    const float* __restrict__ lq2, const float* __restrict__ lk2,
    const float* __restrict__ rw,
    _Float16* __restrict__ Out)        // [b][l][NH*128]
{
    __shared__ _Float16 plds_all[8][16 * 32];

    const int lane = threadIdx.x & 31;
    const int wv   = threadIdx.x >> 5;
    const int l15  = lane & 15;
    const int lhi  = lane >> 4;

    const int wid = blockIdx.x * 8 + wv;
    const int qt  = wid & (LL / 16 - 1);
    int t = wid / (LL / 16);
    const int h = t & (NH - 1);
    const int b = t >> 4;
    const int q0 = qt * 16;

    // lambda_full
    float lsum = 0.f;
    for (int i = lane; i < DD; i += 32) lsum += lq1[i] * lk1[i] + lq2[i] * lk2[i];
    #pragma unroll
    for (int m = 16; m >= 1; m >>= 1) lsum += __shfl_xor(lsum, m, 32);
    const float lam = __expf(lsum) + LAMBDA_INIT;

    const _Float16* Qb = Qh + (size_t)(b * NH + h) * LL * DD;
    const _Float16* Kb = Kh + (size_t)(b * NKV + (h >> 2)) * LL * DD;
    const _Float16* VA = Vt + (size_t)(b * NKV + (h >> 3)) * DD * LL;
    const _Float16* VB = Vt + (size_t)(b * NKV + 2 + (h >> 3)) * DD * LL;

    // Q fragments (A layout): lo = d0..d0+7, hi = d0+8..15 ; d0 = kc*32+lhi*16
    v16h qf[4];
    {
        const _Float16* qr = Qb + (size_t)(q0 + l15) * DD + (lhi << 4);
        #pragma unroll
        for (int kc = 0; kc < 4; ++kc)
            qf[kc] = cat8(ld8(qr + kc * 32), ld8(qr + kc * 32 + 8));
    }

    // streaming per-lane pointers, advanced per 32-key block
    const _Float16* kp0 = Kb + (size_t)l15 * DD + (lhi << 4);          // keys n=l15
    const _Float16* kp1 = kp0 + (size_t)16 * DD;                       // keys n=16+l15
    const _Float16* vpa = VA + (size_t)l15 * LL + (lhi << 4);          // d=l15 (+f*16)
    const _Float16* vpb = VB + (size_t)l15 * LL + (lhi << 4);

    float mrow[8], lrow[8];
    #pragma unroll
    for (int r = 0; r < 8; ++r) { mrow[r] = -3e38f; lrow[r] = 0.f; }
    v8f ofr[16] = {};

    _Float16* pl = &plds_all[wv][0];
    const int rowb = lhi << 3;
    const int nkb = (q0 + 15) / 32 + 1;

    for (int kb = 0; kb < nkb; ++kb) {
        const int kpos0 = kb * 32;

        // prefetch next key block while we chew on this one
        __builtin_prefetch(kp0 + 32 * DD, 0, 1);
        __builtin_prefetch(kp1 + 32 * DD, 0, 1);
        __builtin_prefetch(vpa + 32, 0, 1);
        __builtin_prefetch(vpb + 32, 0, 1);

        // ---- S = Q K^T ----
        v8f s0 = {}, s1 = {};
        #pragma unroll
        for (int kc = 0; kc < 4; ++kc) {
            v16h bf0 = cat8(ld8(kp0 + kc * 32), ld8(kp0 + kc * 32 + 8));
            v16h bf1 = cat8(ld8(kp1 + kc * 32), ld8(kp1 + kc * 32 + 8));
            s0 = WMMA_F16(qf[kc], bf0, s0);
            s1 = WMMA_F16(qf[kc], bf1, s1);
        }

        // ---- causal mask ----
        if (kpos0 + 31 > q0) {
            #pragma unroll
            for (int r = 0; r < 8; ++r) {
                const int row = q0 + rowb + r;
                if (kpos0 + l15 > row)      s0[r] = -1e30f;
                if (kpos0 + 16 + l15 > row) s1[r] = -1e30f;
            }
        }

        // ---- online softmax ----
        float sf[8];
        #pragma unroll
        for (int r = 0; r < 8; ++r) {
            float bmx = fmaxf(s0[r], s1[r]);
            #pragma unroll
            for (int m = 8; m >= 1; m >>= 1) bmx = fmaxf(bmx, __shfl_xor(bmx, m, 32));
            const float mnew = fmaxf(mrow[r], bmx);
            const float scl  = __expf(mrow[r] - mnew);
            mrow[r] = mnew;
            s0[r] = __expf(s0[r] - mnew);
            s1[r] = __expf(s1[r] - mnew);
            float rs = s0[r] + s1[r];
            #pragma unroll
            for (int m = 8; m >= 1; m >>= 1) rs += __shfl_xor(rs, m, 32);
            lrow[r] = lrow[r] * scl + rs;
            sf[r] = scl;
        }
        #pragma unroll
        for (int f = 0; f < 16; ++f)
            #pragma unroll
            for (int r = 0; r < 8; ++r) ofr[f][r] *= sf[r];

        // ---- P (C-layout) -> LDS -> A-fragment ----
        #pragma unroll
        for (int r = 0; r < 8; ++r) {
            pl[(rowb + r) * 32 + l15]      = (_Float16)s0[r];
            pl[(rowb + r) * 32 + 16 + l15] = (_Float16)s1[r];
        }
        asm volatile("s_wait_dscnt 0" ::: "memory");
        const _Float16* pr = pl + l15 * 32 + (lhi << 4);
        v16h pf = cat8(ld8(pr), ld8(pr + 8));
        asm volatile("s_wait_dscnt 0" ::: "memory");

        // ---- O += P @ V_full : key-contiguous loads from Vt ----
        #pragma unroll
        for (int f = 0; f < 8; ++f) {
            v16h bf = cat8(ld8(vpa + (size_t)f * 16 * LL),
                           ld8(vpa + (size_t)f * 16 * LL + 8));
            ofr[f] = WMMA_F16(pf, bf, ofr[f]);
        }
        #pragma unroll
        for (int f = 0; f < 8; ++f) {
            v16h bf = cat8(ld8(vpb + (size_t)f * 16 * LL),
                           ld8(vpb + (size_t)f * 16 * LL + 8));
            ofr[8 + f] = WMMA_F16(pf, bf, ofr[8 + f]);
        }

        kp0 += 32 * DD; kp1 += 32 * DD; vpa += 32; vpb += 32;
    }

    // ---- epilogue: 1/l, diff, RMS norm, *0.8, store f16 ----
    #pragma unroll
    for (int r = 0; r < 8; ++r) {
        const float inv = 1.0f / lrow[r];
        #pragma unroll
        for (int f = 0; f < 16; ++f) ofr[f][r] *= inv;
    }
    float ssq[8];
    #pragma unroll
    for (int r = 0; r < 8; ++r) ssq[r] = 0.f;
    #pragma unroll
    for (int f = 0; f < 8; ++f)
        #pragma unroll
        for (int r = 0; r < 8; ++r) {
            const float d = ofr[f][r] - lam * ofr[f + 8][r];
            ofr[f][r] = d;
            ssq[r] += d * d;
        }
    #pragma unroll
    for (int r = 0; r < 8; ++r) {
        float s = ssq[r];
        #pragma unroll
        for (int m = 8; m >= 1; m >>= 1) s += __shfl_xor(s, m, 32);
        ssq[r] = rsqrtf(s * (1.0f / 128.0f) + 1e-6f) * (1.0f - LAMBDA_INIT);
    }
    #pragma unroll
    for (int f = 0; f < 8; ++f) {
        const float w = rw[f * 16 + l15];
        #pragma unroll
        for (int r = 0; r < 8; ++r) {
            const int row = q0 + rowb + r;
            Out[(size_t)(b * LL + row) * (NH * DD) + h * DD + f * 16 + l15] =
                (_Float16)(ofr[f][r] * ssq[r] * w);
        }
    }
}

// ---------------------------------------------------------------------------
// host launcher
// ---------------------------------------------------------------------------
extern "C" void kernel_launch(void* const* d_in, const int* in_sizes, int n_in,
                              void* d_out, int out_size, void* d_ws, size_t ws_size,
                              hipStream_t stream)
{
    const float* hs  = (const float*)d_in[0];
    const float* cosb= (const float*)d_in[1];
    const float* sinb= (const float*)d_in[2];
    const float* Wq  = (const float*)d_in[3];
    const float* Wk  = (const float*)d_in[4];
    const float* Wv  = (const float*)d_in[5];
    const float* Wo  = (const float*)d_in[6];
    const float* lq1 = (const float*)d_in[7];
    const float* lk1 = (const float*)d_in[8];
    const float* lq2 = (const float*)d_in[9];
    const float* lk2 = (const float*)d_in[10];
    const float* rw  = (const float*)d_in[11];
    float* out = (float*)d_out;

    const int M = BB * LL;   // 4096
    const int MB = 1 << 20;
    char* ws = (char*)d_ws;
    _Float16* hsh = (_Float16*)(ws + 0);        // 16 MiB  [4096][2048] f16
    _Float16* Wqt = (_Float16*)(ws + 16 * MB);  //  8 MiB  [2048][2048]
    _Float16* Wkt = (_Float16*)(ws + 24 * MB);  //  2 MiB  [512][2048]
    _Float16* Wvt = (_Float16*)(ws + 26 * MB);  //  2 MiB
    _Float16* Wot = (_Float16*)(ws + 28 * MB);  //  8 MiB  [2048][2048]
    float*    Qf  = (float*)   (ws + 36 * MB);  // 32 MiB  [4096][2048] f32
    float*    Kf  = (float*)   (ws + 68 * MB);  //  8 MiB  [4096][512]
    float*    Vf  = (float*)   (ws + 76 * MB);  //  8 MiB
    _Float16* Qh  = (_Float16*)(ws + 84 * MB);  // 16 MiB  [b][16][L][128]
    _Float16* Kh  = (_Float16*)(ws + 100 * MB); //  4 MiB  [b][4][L][128]
    _Float16* Vt  = (_Float16*)(ws + 104 * MB); //  4 MiB  [b][4][128][L]
    _Float16* Ah  = (_Float16*)(ws + 36 * MB);  // reuse Qf region (dead by then)

    // 0) one-shot converts: hs->f16, weights -> f16 transposed
    f32_to_f16_kernel<<<(M * HID) / 256, 256, 0, stream>>>(hs, hsh);
    wconv_t_kernel<<<(HID * HID) / 256, 256, 0, stream>>>(Wq, Wqt, HID, HID);
    wconv_t_kernel<<<(HID * NKV * DD) / 256, 256, 0, stream>>>(Wk, Wkt, HID, NKV * DD);
    wconv_t_kernel<<<(HID * NKV * DD) / 256, 256, 0, stream>>>(Wv, Wvt, HID, NKV * DD);
    wconv_t_kernel<<<(HID * HID) / 256, 256, 0, stream>>>(Wo, Wot, HID, HID);

    // 1) projections
    gemm_f16t_kernel<<<dim3(HID / 128, M / 128), 256, 0, stream>>>(hsh, Wqt, Qf, M, HID, HID);
    gemm_f16t_kernel<<<dim3((NKV * DD) / 128, M / 128), 256, 0, stream>>>(hsh, Wkt, Kf, M, NKV * DD, HID);
    gemm_f16t_kernel<<<dim3((NKV * DD) / 128, M / 128), 256, 0, stream>>>(hsh, Wvt, Vf, M, NKV * DD, HID);

    // 2) RoPE (+1/sqrt(D) on Q) and V transpose
    const float qscale = 0.08838834764831845f;
    rope_transpose_kernel<<<(BB * LL * NH * 64) / 256, 256, 0, stream>>>(Qf, cosb, sinb, qscale, NH, Qh);
    rope_transpose_kernel<<<(BB * LL * NKV * 64) / 256, 256, 0, stream>>>(Kf, cosb, sinb, 1.0f, NKV, Kh);
    vconv_t_kernel<<<(BB * LL * NKV * DD) / 256, 256, 0, stream>>>(Vf, Vt);

    // 3) flash differential attention -> Ah f16 [b][l][2048]
    attn_kernel<<<(BB * NH * (LL / 16)) / 8, 256, 0, stream>>>(Qh, Kh, Vt, lq1, lk1, lq2, lk2, rw, Ah);

    // 4) output projection
    gemm_f16t_kernel<<<dim3(HID / 128, M / 128), 256, 0, stream>>>(Ah, Wot, out, M, HID, HID);
}
